// MixtureOfExperts_66967130079474
// MI455X (gfx1250) — matrix operand order repaired and checked by reading, not compiled, not run on hardware
//
#include <hip/hip_runtime.h>
#include <cmath>

#define Hdim 1024
#define Idim 1408
#define Edim 8
#define Ttok 2048

#define BM 128
#define BN 64
#define BK 32
#define LP 40    // LDS row pitch (elements): 80 B = 20 banks -> conflict-free rows

typedef __attribute__((ext_vector_type(16))) __bf16 v16bf;
typedef __attribute__((ext_vector_type(8)))  __bf16 v8bf;
typedef __attribute__((ext_vector_type(8)))  float  v8f;

union ABfrag { v16bf v; v8bf h[2]; };

// ---------------- async global->LDS path (gfx1250), guarded ----------------
#if defined(__has_builtin)
#if __has_builtin(__builtin_amdgcn_global_load_async_to_lds_b128)
#define USE_ASYNC_LDS 1
#endif
#endif
#ifndef USE_ASYNC_LDS
#define USE_ASYNC_LDS 0
#endif

#if USE_ASYNC_LDS
typedef int v4i_ __attribute__((vector_size(16)));
typedef __attribute__((address_space(1))) v4i_ g_v4i;   // global b128 element
typedef __attribute__((address_space(3))) v4i_ l_v4i;   // LDS b128 element

__device__ __forceinline__ void async_b128(const unsigned short* g, unsigned short* l) {
  __builtin_amdgcn_global_load_async_to_lds_b128(
      (g_v4i*)(unsigned long long)(size_t)g,
      (l_v4i*)(unsigned)(size_t)l, 0, 0);
}
#endif

__device__ __forceinline__ void wait_async() {
#if USE_ASYNC_LDS
#if __has_builtin(__builtin_amdgcn_s_wait_asynccnt)
  __builtin_amdgcn_s_wait_asynccnt(0);
#else
  asm volatile("s_wait_asynccnt 0x0" ::: "memory");
#endif
#endif
}

__device__ __forceinline__ unsigned short f2bf(float f) {
  union { float f; unsigned u; } v; v.f = f;
  unsigned r = v.u + 0x7FFFu + ((v.u >> 16) & 1u);   // round-to-nearest-even
  return (unsigned short)(r >> 16);
}

// packed f32->bf16: hardware cvt if declared, else bias + single v_perm_b32
__device__ __forceinline__ unsigned pack_bf16(float lo, float hi) {
#if defined(__has_builtin) && __has_builtin(__builtin_amdgcn_cvt_pk_bf16_f32)
  auto p = __builtin_amdgcn_cvt_pk_bf16_f32(lo, hi);
  unsigned u; __builtin_memcpy(&u, &p, 4); return u;
#else
  union { float f; unsigned u; } a, b;
  a.f = lo; b.f = hi;
  // round-half-up bias, then take both high halves in one v_perm_b32
  return __builtin_amdgcn_perm(b.u + 0x8000u, a.u + 0x8000u, 0x07060302u);
#endif
}

__device__ __forceinline__ uint4 cvt8_f32_bf16(float4 a, float4 b) {
  uint4 r;
  r.x = pack_bf16(a.x, a.y);
  r.y = pack_bf16(a.z, a.w);
  r.z = pack_bf16(b.x, b.y);
  r.w = pack_bf16(b.z, b.w);
  return r;
}

// ---------------- utility kernels ----------------

__global__ void zero_f32_kernel(float* __restrict__ p, int n) {
  int i = blockIdx.x * blockDim.x + threadIdx.x;
  if (i < n) p[i] = 0.0f;
}

__global__ void cvt_f32_bf16_kernel(const float* __restrict__ src,
                                    unsigned short* __restrict__ dst, int n) {
  int i = blockIdx.x * blockDim.x + threadIdx.x;
  int stride = gridDim.x * blockDim.x;
  for (; i < n; i += stride) dst[i] = f2bf(src[i]);
}

// ---------------- router: one wave32 per token ----------------
// gate[t,e] = raw logit if e in top-2 of logits else 0  (ROUTING_SCALE = 1)

__global__ void __launch_bounds__(256)
router_gate_kernel(const float* __restrict__ x, const float* __restrict__ rw,
                   float* __restrict__ gate) {
  int wave = threadIdx.x >> 5;
  int lane = threadIdx.x & 31;
  int t = blockIdx.x * (blockDim.x >> 5) + wave;
  if (t >= Ttok) return;

  float acc[Edim];
#pragma unroll
  for (int e = 0; e < Edim; ++e) acc[e] = 0.0f;

  for (int k = lane; k < Hdim; k += 32) {
    float xv = x[(size_t)t * Hdim + k];
#pragma unroll
    for (int e = 0; e < Edim; ++e) acc[e] += xv * rw[e * Hdim + k];
  }
#pragma unroll
  for (int e = 0; e < Edim; ++e) {
#pragma unroll
    for (int off = 16; off > 0; off >>= 1)
      acc[e] += __shfl_xor(acc[e], off, 32);
  }
  if (lane == 0) {
    int i1 = 0; float v1 = acc[0];
    for (int e = 1; e < Edim; ++e) if (acc[e] > v1) { v1 = acc[e]; i1 = e; }
    int i2 = -1; float v2 = -3.0e38f;
    for (int e = 0; e < Edim; ++e)
      if (e != i1 && acc[e] > v2) { v2 = acc[e]; i2 = e; }
    for (int e = 0; e < Edim; ++e)
      gate[(size_t)t * Edim + e] = (e == i1 || e == i2) ? acc[e] : 0.0f;
  }
}

// ---------------- up-projection + SwiGLU (WMMA bf16, double-buffered) --------
// hb[T, I] = silu(xb @ w1^T) * (xb @ w3^T);  w1/w3 fp32 [I, H], converted in-kernel.

__global__ void __launch_bounds__(256)
gemm_swiglu_kernel(const unsigned short* __restrict__ xb,  // [T, H] bf16
                   const float* __restrict__ w1,           // [I, H] fp32
                   const float* __restrict__ w3,           // [I, H] fp32
                   unsigned short* __restrict__ hb)        // [T, I] bf16
{
  __shared__ alignas(64) unsigned short Xs[2][BM * LP];
  __shared__ alignas(64) unsigned short W1s[2][BN * LP];
  __shared__ alignas(64) unsigned short W3s[2][BN * LP];

  const int m0 = blockIdx.x * BM;
  const int n0 = blockIdx.y * BN;
  const int wid  = threadIdx.x >> 5;
  const int lane = threadIdx.x & 31;
  const int wm = wid & 3;        // 4 waves along M (32 rows each)
  const int wn = wid >> 2;       // 2 waves along N (32 cols each)
  const int lgrp = lane >> 4;    // K-half selector
  const int l16  = lane & 15;

  v8f acc1[2][2], acc3[2][2];
  v8f zer = {0.f, 0.f, 0.f, 0.f, 0.f, 0.f, 0.f, 0.f};
#pragma unroll
  for (int a = 0; a < 2; ++a)
#pragma unroll
    for (int b = 0; b < 2; ++b) { acc1[a][b] = zer; acc3[a][b] = zer; }

  const int xr = threadIdx.x >> 1, xh = threadIdx.x & 1;  // X tile: 128 rows x 2 halves
  const int wr = threadIdx.x >> 2, wq = threadIdx.x & 3;  // W tile: 64 rows x 4 octets

  const unsigned short* xg = xb + (size_t)(m0 + xr) * Hdim + xh * 16;
  const float* w1g = w1 + (size_t)(n0 + wr) * Hdim + wq * 8;
  const float* w3g = w3 + (size_t)(n0 + wr) * Hdim + wq * 8;
  const int xOff = xr * LP + xh * 16;
  const int wOff = wr * LP + wq * 8;

  auto stage = [&](int buf, int k0) {
#if USE_ASYNC_LDS
    async_b128(xg + k0, &Xs[buf][xOff]);
    async_b128(xg + k0 + 8, &Xs[buf][xOff] + 8);
#else
    const uint4* s = (const uint4*)(const void*)(xg + k0);
    uint4* d = (uint4*)(void*)&Xs[buf][xOff];
    d[0] = s[0]; d[1] = s[1];
#endif
    const float4* s1 = (const float4*)(const void*)(w1g + k0);
    const float4* s3 = (const float4*)(const void*)(w3g + k0);
    *(uint4*)(void*)&W1s[buf][wOff] = cvt8_f32_bf16(s1[0], s1[1]);
    *(uint4*)(void*)&W3s[buf][wOff] = cvt8_f32_bf16(s3[0], s3[1]);
    if (k0 + BK < Hdim) {   // global_prefetch_b8 for the following tile
      __builtin_prefetch(w1g + k0 + BK, 0, 1);
      __builtin_prefetch(w3g + k0 + BK, 0, 1);
    }
  };

  auto compute = [&](int buf) {
    ABfrag a[2], b1[2], b3[2];
#pragma unroll
    for (int tm = 0; tm < 2; ++tm) {
      int row = wm * 32 + tm * 16 + l16;
      a[tm].h[0] = *(const v8bf*)(const void*)(&Xs[buf][row * LP + lgrp * 16]);
      a[tm].h[1] = *(const v8bf*)(const void*)(&Xs[buf][row * LP + lgrp * 16 + 8]);
    }
#pragma unroll
    for (int tn = 0; tn < 2; ++tn) {
      int col = wn * 32 + tn * 16 + l16;
      b1[tn].h[0] = *(const v8bf*)(const void*)(&W1s[buf][col * LP + lgrp * 16]);
      b1[tn].h[1] = *(const v8bf*)(const void*)(&W1s[buf][col * LP + lgrp * 16 + 8]);
      b3[tn].h[0] = *(const v8bf*)(const void*)(&W3s[buf][col * LP + lgrp * 16]);
      b3[tn].h[1] = *(const v8bf*)(const void*)(&W3s[buf][col * LP + lgrp * 16 + 8]);
    }
#pragma unroll
    for (int tm = 0; tm < 2; ++tm)
#pragma unroll
      for (int tn = 0; tn < 2; ++tn) {
        acc1[tm][tn] = __builtin_amdgcn_wmma_f32_16x16x32_bf16(
            false, a[tm].v, false, b1[tn].v, (short)0, acc1[tm][tn], false, false);
        acc3[tm][tn] = __builtin_amdgcn_wmma_f32_16x16x32_bf16(
            false, a[tm].v, false, b3[tn].v, (short)0, acc3[tm][tn], false, false);
      }
  };

  const int nit = Hdim / BK;   // 32, even
  stage(0, 0);
  wait_async();
  __syncthreads();
  for (int it = 0; it < nit; it += 2) {
    // even step: compute buf0 while buf1 fills
    if (it + 1 < nit) stage(1, (it + 1) * BK);
    compute(0);
    wait_async();
    __syncthreads();
    // odd step: compute buf1 while buf0 fills
    if (it + 2 < nit) stage(0, (it + 2) * BK);
    compute(1);
    wait_async();
    __syncthreads();
  }

  // epilogue: h = silu(acc1) * acc3 -> bf16
#pragma unroll
  for (int tm = 0; tm < 2; ++tm)
#pragma unroll
    for (int tn = 0; tn < 2; ++tn) {
      int col = n0 + wn * 32 + tn * 16 + l16;
      int rowBase = m0 + wm * 32 + tm * 16 + lgrp * 8;
#pragma unroll
      for (int i = 0; i < 8; ++i) {
        float g = acc1[tm][tn][i];
        float s = g / (1.0f + __expf(-g));
        float hv = s * acc3[tm][tn][i];
        hb[(size_t)(rowBase + i) * Idim + col] = f2bf(hv);
      }
    }
}

// ---------------- down-projection with gated accumulation --------------------
// out[T, H] += gate[:,e] * (hb @ w2^T);  w2 fp32 [H, I], converted in-kernel.

__global__ void __launch_bounds__(256)
gemm_down_kernel(const unsigned short* __restrict__ hbuf,  // [T, I] bf16
                 const float* __restrict__ w2,             // [H, I] fp32
                 const float* __restrict__ gate,           // [T, E] or nullptr
                 int expert,
                 float* __restrict__ out)                  // [T, H] fp32 accum
{
  __shared__ alignas(64) unsigned short As[2][BM * LP];
  __shared__ alignas(64) unsigned short Bs[2][BN * LP];

  const int m0 = blockIdx.x * BM;
  const int n0 = blockIdx.y * BN;
  const int wid  = threadIdx.x >> 5;
  const int lane = threadIdx.x & 31;
  const int wm = wid & 3;
  const int wn = wid >> 2;
  const int lgrp = lane >> 4;
  const int l16  = lane & 15;

  v8f acc[2][2];
  v8f zer = {0.f, 0.f, 0.f, 0.f, 0.f, 0.f, 0.f, 0.f};
#pragma unroll
  for (int a = 0; a < 2; ++a)
#pragma unroll
    for (int b = 0; b < 2; ++b) acc[a][b] = zer;

  const int xr = threadIdx.x >> 1, xh = threadIdx.x & 1;
  const int wr = threadIdx.x >> 2, wq = threadIdx.x & 3;

  const unsigned short* ag = hbuf + (size_t)(m0 + xr) * Idim + xh * 16;
  const float* bg = w2 + (size_t)(n0 + wr) * Idim + wq * 8;
  const int aOff = xr * LP + xh * 16;
  const int bOff = wr * LP + wq * 8;

  auto stage = [&](int buf, int k0) {
#if USE_ASYNC_LDS
    async_b128(ag + k0, &As[buf][aOff]);
    async_b128(ag + k0 + 8, &As[buf][aOff] + 8);
#else
    const uint4* s = (const uint4*)(const void*)(ag + k0);
    uint4* d = (uint4*)(void*)&As[buf][aOff];
    d[0] = s[0]; d[1] = s[1];
#endif
    const float4* s2 = (const float4*)(const void*)(bg + k0);
    *(uint4*)(void*)&Bs[buf][bOff] = cvt8_f32_bf16(s2[0], s2[1]);
    if (k0 + BK < Idim)
      __builtin_prefetch(bg + k0 + BK, 0, 1);
  };

  auto compute = [&](int buf) {
    ABfrag a[2], b[2];
#pragma unroll
    for (int tm = 0; tm < 2; ++tm) {
      int row = wm * 32 + tm * 16 + l16;
      a[tm].h[0] = *(const v8bf*)(const void*)(&As[buf][row * LP + lgrp * 16]);
      a[tm].h[1] = *(const v8bf*)(const void*)(&As[buf][row * LP + lgrp * 16 + 8]);
    }
#pragma unroll
    for (int tn = 0; tn < 2; ++tn) {
      int col = wn * 32 + tn * 16 + l16;
      b[tn].h[0] = *(const v8bf*)(const void*)(&Bs[buf][col * LP + lgrp * 16]);
      b[tn].h[1] = *(const v8bf*)(const void*)(&Bs[buf][col * LP + lgrp * 16 + 8]);
    }
#pragma unroll
    for (int tm = 0; tm < 2; ++tm)
#pragma unroll
      for (int tn = 0; tn < 2; ++tn)
        acc[tm][tn] = __builtin_amdgcn_wmma_f32_16x16x32_bf16(
            false, a[tm].v, false, b[tn].v, (short)0, acc[tm][tn], false, false);
  };

  const int nit = Idim / BK;   // 44, even
  stage(0, 0);
  wait_async();
  __syncthreads();
  for (int it = 0; it < nit; it += 2) {
    if (it + 1 < nit) stage(1, (it + 1) * BK);
    compute(0);
    wait_async();
    __syncthreads();
    if (it + 2 < nit) stage(0, (it + 2) * BK);
    compute(1);
    wait_async();
    __syncthreads();
  }

#pragma unroll
  for (int tm = 0; tm < 2; ++tm)
#pragma unroll
    for (int tn = 0; tn < 2; ++tn) {
      int col = n0 + wn * 32 + tn * 16 + l16;
      int rowBase = m0 + wm * 32 + tm * 16 + lgrp * 8;
#pragma unroll
      for (int i = 0; i < 8; ++i) {
        int row = rowBase + i;
        float g = gate ? gate[(size_t)row * Edim + expert] : 1.0f;
        if (g != 0.0f)
          out[(size_t)row * Hdim + col] += g * acc[tm][tn][i];
      }
    }
}

// ---------------- launcher ----------------

extern "C" void kernel_launch(void* const* d_in, const int* in_sizes, int n_in,
                              void* d_out, int out_size, void* d_ws, size_t ws_size,
                              hipStream_t stream) {
  (void)in_sizes; (void)n_in; (void)out_size; (void)ws_size;

  const float* x   = (const float*)d_in[0];
  const float* rw  = (const float*)d_in[1];
  const float* w1  = (const float*)d_in[2];
  const float* w2  = (const float*)d_in[3];
  const float* w3  = (const float*)d_in[4];
  const float* sw1 = (const float*)d_in[5];
  const float* sw2 = (const float*)d_in[6];
  const float* sw3 = (const float*)d_in[7];
  float* out = (float*)d_out;

  char* ws = (char*)d_ws;
  unsigned short* xb = (unsigned short*)ws; ws += (size_t)Ttok * Hdim * 2;  // 4 MB
  unsigned short* hb = (unsigned short*)ws; ws += (size_t)Ttok * Idim * 2;  // 5.77 MB
  float* gate = (float*)ws;                                                 // 64 KB

  const int nOut = Ttok * Hdim;
  zero_f32_kernel<<<(nOut + 255) / 256, 256, 0, stream>>>(out, nOut);

  cvt_f32_bf16_kernel<<<2048, 256, 0, stream>>>(x, xb, Ttok * Hdim);

  router_gate_kernel<<<Ttok / 8, 256, 0, stream>>>(x, rw, gate);

  dim3 gridUp(Ttok / BM, Idim / BN);    // 16 x 22
  dim3 gridDn(Ttok / BM, Hdim / BN);    // 16 x 16
  const int wElems = Idim * Hdim;       // 1,441,792

  for (int e = 0; e < Edim + 1; ++e) {
    const float* W1 = (e < Edim) ? (w1 + (size_t)e * wElems) : sw1;
    const float* W3 = (e < Edim) ? (w3 + (size_t)e * wElems) : sw3;
    const float* W2 = (e < Edim) ? (w2 + (size_t)e * wElems) : sw2;

    gemm_swiglu_kernel<<<gridUp, 256, 0, stream>>>(xb, W1, W3, hb);
    gemm_down_kernel<<<gridDn, 256, 0, stream>>>(
        hb, W2, (e < Edim) ? gate : nullptr, e, out);
  }
}
